// GraphConvoNetwork_6940667150636
// MI455X (gfx1250) — compile-verified, matrix operand order checked by reading, not run on hardware
//
#include <hip/hip_runtime.h>

#define N_NODES 100000
#define N_EDGES 1600000
#define FDIM 128
#define HDIM 128
#define ODIM 64
#define NPAD 100096   /* 128 * 782, multiple of 16 for WMMA tiles */
#define BN_EPS 1e-5f

typedef __attribute__((ext_vector_type(16))) __bf16 v16bf;
typedef __attribute__((ext_vector_type(8)))  float  v8f;

// ---------- helpers ----------
__device__ __forceinline__ unsigned short f2bf_bits(float f) {
    unsigned u = __builtin_bit_cast(unsigned, f);
    unsigned r = (u + 0x7FFFu + ((u >> 16) & 1u)) >> 16;   // RNE
    return (unsigned short)r;
}

// K-within-tile index for the 16-bit 16x32 A / 32x16 B WMMA operand layout:
// VGPR v in 0..3 holds K = 8*half + 2v + p ; v in 4..7 holds K = 16 + 8*half + 2(v-4) + p
__device__ __forceinline__ int klocal(int j, int half) {
    int v = j >> 1, p = j & 1;
    return (v < 4) ? (8 * half + 2 * v + p) : (16 + 8 * half + 2 * (v - 4) + p);
}

// ---------- weight pre-pack into per-lane-contiguous bf16 B fragments ----------
// frag[(kt*nct + ct)*32*16 + lane*16 + j] = bf16( W[k, n] ), row-major W [K, Ncols]
__global__ void pack_w_kernel(const float* __restrict__ W, unsigned short* __restrict__ frag,
                              int K, int Ncols) {
    int nct = Ncols >> 4;
    int total = (K >> 5) * nct * 512;
    int gid = blockIdx.x * blockDim.x + threadIdx.x;
    if (gid >= total) return;
    int j    = gid & 15;
    int lane = (gid >> 4) & 31;
    int rem  = gid >> 9;
    int ct = rem % nct;
    int kt = rem / nct;
    int half = lane >> 4;
    int n = ct * 16 + (lane & 15);
    int k = kt * 32 + klocal(j, half);
    frag[gid] = f2bf_bits(W[(size_t)k * Ncols + n]);
}

// ---------- degree / dinv ----------
__global__ void degree_kernel(const int* __restrict__ ei, float* __restrict__ deg) {
    int e = blockIdx.x * blockDim.x + threadIdx.x;
    if (e >= N_EDGES) return;
    unsafeAtomicAdd(&deg[ei[e]], 1.0f);           // row = ei[0][e]
}

__global__ void dinv_kernel(float* __restrict__ d) {
    int i = blockIdx.x * blockDim.x + threadIdx.x;
    if (i >= NPAD) return;
    d[i] = (i < N_NODES) ? rsqrtf(d[i] + 1.0f) : 0.0f;   // +1 for self loop; always > 0
}

// ---------- tiled WMMA GEMM:  C[rt*16.., ct*16..] = A(16x128) * Wfrag ----------
// blockDim.x = 32 * (Ncols/16); one wave per 16-col tile; K fixed at 128.
__global__ void gemm_wmma_kernel(const float* __restrict__ A,
                                 const unsigned short* __restrict__ wfrag,
                                 const float* __restrict__ bias,
                                 float* __restrict__ C,
                                 int Mvalid, int Mstore, int Ncols) {
    __shared__ unsigned short lds_a[16 * 128];
    const int K = 128;
    int nct = Ncols >> 4;
    int rt = blockIdx.x;

    // stage 16x128 f32 tile of A into LDS as bf16 (zero-fill past Mvalid)
    for (int idx = threadIdx.x; idx < 16 * 128; idx += blockDim.x) {
        int m = idx >> 7, k = idx & 127;
        int row = rt * 16 + m;
        float v = (row < Mvalid) ? A[(size_t)row * K + k] : 0.0f;
        lds_a[idx] = f2bf_bits(v);
    }
    __syncthreads();

    int lane = threadIdx.x & 31;
    int ct   = threadIdx.x >> 5;
    int half = lane >> 4;
    int mloc = lane & 15;

    v8f acc = {0.f, 0.f, 0.f, 0.f, 0.f, 0.f, 0.f, 0.f};
    for (int kt = 0; kt < K / 32; ++kt) {
        v16bf a;
#pragma unroll
        for (int j = 0; j < 16; ++j) {
            unsigned short bits = lds_a[mloc * 128 + kt * 32 + klocal(j, half)];
            a[j] = __builtin_bit_cast(__bf16, bits);
        }
        v16bf b = *reinterpret_cast<const v16bf*>(
            wfrag + ((size_t)(kt * nct + ct) * 32 + lane) * 16);
        acc = __builtin_amdgcn_wmma_f32_16x16x32_bf16(
            /*neg_a=*/false, a, /*neg_b=*/false, b,
            /*c_mod=*/(short)0, acc, /*reuse_a=*/false, /*reuse_b=*/false);
    }

    // D layout: VGPR v -> M = v + 8*half ; N = lane%16
    int col = ct * 16 + mloc;
    float bv = bias ? bias[col] : 0.0f;
    int row0 = rt * 16 + 8 * half;           // this lane's 8 rows are row0..row0+7
    if (rt * 16 + 16 <= Mstore) {
        // full tile (always true for padded GEMMs): branch-free pointer-bump stores
        float* p = C + (size_t)row0 * Ncols + col;
#pragma unroll
        for (int v = 0; v < 8; ++v) {
            *p = acc[v] + bv;
            p += Ncols;
        }
    } else {
#pragma unroll
        for (int v = 0; v < 8; ++v) {
            int row = row0 + v;
            if (row < Mstore) C[(size_t)row * Ncols + col] = acc[v] + bv;
        }
    }
}

// ---------- edge aggregation: one wave per edge, float4 gather + f32 atomic scatter ----------
__global__ void edge_agg_kernel(const int* __restrict__ ei, const float* __restrict__ dinv,
                                const float* __restrict__ h, float* __restrict__ out) {
    int lane = threadIdx.x & 31;
    int w    = threadIdx.x >> 5;
    long long e = (long long)blockIdx.x * 8 + w;
    if (e >= N_EDGES) return;
    int r = ei[e];
    int c = ei[N_EDGES + e];
    float nrm = dinv[r] * dinv[c];
    float4 v = reinterpret_cast<const float4*>(h + (size_t)c * HDIM)[lane];
    float* o = out + (size_t)r * HDIM + lane * 4;
    unsafeAtomicAdd(o + 0, v.x * nrm);
    unsafeAtomicAdd(o + 1, v.y * nrm);
    unsafeAtomicAdd(o + 2, v.z * nrm);
    unsafeAtomicAdd(o + 3, v.w * nrm);
}

// ---------- self-loop + bias (+optional ReLU) ----------
__global__ void selfloop_bias_kernel(float* __restrict__ out, const float* __restrict__ h,
                                     const float* __restrict__ dinv,
                                     const float* __restrict__ bias, int do_relu) {
    size_t gid = (size_t)blockIdx.x * blockDim.x + threadIdx.x;
    if (gid >= (size_t)N_NODES * HDIM) return;
    int i = (int)(gid >> 7);
    int f = (int)(gid & 127);
    float d = dinv[i];
    float v = out[gid] + h[gid] * d * d + bias[f];
    if (do_relu) v = v > 0.f ? v : 0.f;
    out[gid] = v;
}

// ---------- BatchNorm: per-feature sums (atomic) then normalize+ReLU ----------
__global__ void bn_stats_kernel(const float* __restrict__ buf, float* __restrict__ stats) {
    int f  = threadIdx.x;                 // 128 threads
    int r0 = blockIdx.x * 256;
    float s = 0.f, ss = 0.f;
    for (int i = 0; i < 256; ++i) {
        int r = r0 + i;
        if (r < N_NODES) {
            float v = buf[(size_t)r * HDIM + f];
            s += v; ss += v * v;
        }
    }
    unsafeAtomicAdd(&stats[f], s);
    unsafeAtomicAdd(&stats[128 + f], ss);
}

__global__ void bn_relu_kernel(float* __restrict__ buf, const float* __restrict__ stats,
                               const float* __restrict__ gamma, const float* __restrict__ beta) {
    size_t gid = (size_t)blockIdx.x * blockDim.x + threadIdx.x;
    if (gid >= (size_t)N_NODES * HDIM) return;
    int f = (int)(gid & 127);
    const float invn = 1.0f / (float)N_NODES;
    float mean  = stats[f] * invn;
    float var   = stats[128 + f] * invn - mean * mean;
    float scale = gamma[f] * rsqrtf(var + BN_EPS);
    float y = (buf[gid] - mean) * scale + beta[f];
    buf[gid] = y > 0.f ? y : 0.f;
}

// ---------- launch ----------
extern "C" void kernel_launch(void* const* d_in, const int* in_sizes, int n_in,
                              void* d_out, int out_size, void* d_ws, size_t ws_size,
                              hipStream_t stream) {
    const float* x      = (const float*)d_in[0];
    const int*   ei     = (const int*)d_in[1];     // [2, E] int32
    const float* W1     = (const float*)d_in[2];
    const float* b1     = (const float*)d_in[3];
    const float* gamma1 = (const float*)d_in[4];
    const float* beta1  = (const float*)d_in[5];
    const float* W2     = (const float*)d_in[6];
    const float* b2     = (const float*)d_in[7];
    const float* W3     = (const float*)d_in[8];
    const float* b3     = (const float*)d_in[9];
    float* out = (float*)d_out;

    char* ws = (char*)d_ws;
    const size_t SZ    = (size_t)NPAD * HDIM * sizeof(float);   // 51.25 MB, 256B aligned
    const size_t dinvB = (size_t)NPAD * sizeof(float);
    float* bufA  = (float*)(ws);
    float* bufB  = (float*)(ws + SZ);
    float* dinv  = (float*)(ws + 2 * SZ);
    float* stats = (float*)(ws + 2 * SZ + dinvB);
    unsigned short* wf1 = (unsigned short*)(ws + 2 * SZ + dinvB + 1024);
    unsigned short* wf2 = wf1 + 128 * 128;
    unsigned short* wf3 = wf2 + 128 * 128;

    // pack weights into WMMA B-fragment order (bf16)
    pack_w_kernel<<<(128 * 128 + 255) / 256, 256, 0, stream>>>(W1, wf1, 128, 128);
    pack_w_kernel<<<(128 * 128 + 255) / 256, 256, 0, stream>>>(W2, wf2, 128, 128);
    pack_w_kernel<<<(128 * 64  + 255) / 256, 256, 0, stream>>>(W3, wf3, 128, 64);

    // degrees -> dinv
    hipMemsetAsync(dinv, 0, dinvB, stream);
    degree_kernel<<<(N_EDGES + 255) / 256, 256, 0, stream>>>(ei, dinv);
    dinv_kernel<<<(NPAD + 255) / 256, 256, 0, stream>>>(dinv);

    const int rowTiles = NPAD / 16;
    const int elemBlocks = (int)(((size_t)N_NODES * HDIM + 255) / 256);

    // ----- layer 1 -----
    gemm_wmma_kernel<<<rowTiles, 256, 0, stream>>>(x, wf1, nullptr, bufA, N_NODES, NPAD, 128);
    hipMemsetAsync(bufB, 0, SZ, stream);
    edge_agg_kernel<<<(N_EDGES + 7) / 8, 256, 0, stream>>>(ei, dinv, bufA, bufB);
    selfloop_bias_kernel<<<elemBlocks, 256, 0, stream>>>(bufB, bufA, dinv, b1, 0);
    hipMemsetAsync(stats, 0, 256 * sizeof(float), stream);
    bn_stats_kernel<<<(N_NODES + 255) / 256, 128, 0, stream>>>(bufB, stats);
    bn_relu_kernel<<<elemBlocks, 256, 0, stream>>>(bufB, stats, gamma1, beta1);

    // ----- layer 2 -----
    gemm_wmma_kernel<<<rowTiles, 256, 0, stream>>>(bufB, wf2, nullptr, bufA, NPAD, NPAD, 128);
    hipMemsetAsync(bufB, 0, SZ, stream);
    edge_agg_kernel<<<(N_EDGES + 7) / 8, 256, 0, stream>>>(ei, dinv, bufA, bufB);
    selfloop_bias_kernel<<<elemBlocks, 256, 0, stream>>>(bufB, bufA, dinv, b2, 1);

    // ----- head: [N,128] @ [128,64] + b3 -> d_out -----
    gemm_wmma_kernel<<<rowTiles, 128, 0, stream>>>(bufB, wf3, b3, out, NPAD, N_NODES, 64);
}
